// SelfAttention_20933670601106
// MI455X (gfx1250) — compile-verified, matrix operand order checked by reading, not run on hardware
//
#include <hip/hip_runtime.h>

// ---------------------------------------------------------------------------
// Fused self-attention for MI455X (gfx1250, wave32, WMMA bf16 -> f32 acc)
// S=2048, B=2, E=1024, H=16, DK=DV=64
// Round 2: 4-wave workgroups, K/V/B panels staged to LDS with
// global_load_async_to_lds_b128 (ASYNCcnt), double-buffered against WMMAs.
//
// Workspace layout (bytes), total 48 MB:
//   [ 0,  8M) xb   : X as bf16, row-major [4096][1024]
//   [ 8M,10M) wqt  : Wq^T bf16 [1024][1024]  (N-major -> B-operand == A pattern)
//   [10M,12M) wkt  : Wk^T bf16
//   [12M,14M) wvt  : Wv^T bf16
//   [14M,16M) wot  : Wo^T bf16
//   [16M,24M) qbuf : Q  bf16 [B][H][S][DK]
//   [24M,32M) kbuf : K  bf16 [B][H][S][DK]
//   [32M,40M) vtb  : V^T bf16 [B][H][DV][S]
//   [40M,48M) ctx  : ctx bf16 [(s*B+b)][H*DV]
// ---------------------------------------------------------------------------

typedef __bf16 bf16_t;
typedef __attribute__((ext_vector_type(16))) __bf16 v16bf;
typedef __attribute__((ext_vector_type(8)))  __bf16 v8bf;
typedef __attribute__((ext_vector_type(8)))  float  v8f;

#define S_LEN 2048
#define BATCH 2
#define EMB   1024
#define HEADS 16
#define DKV   64
#define ROWS  (S_LEN * BATCH)   // 4096

static __device__ __forceinline__ v8f wmma_bf16(v16bf a, v16bf b, v8f c) {
  return __builtin_amdgcn_wmma_f32_16x16x32_bf16(
      /*neg_a=*/false, a, /*neg_b=*/false, b,
      /*c_mod=*/(short)0, c, /*reuse_a=*/false, /*reuse_b=*/false);
}

// 16-bit A-operand (and B-operand; lane==N) loader per CDNA5 ISA layout:
// lane&15 selects the row; half=lane>>4 selects K-subchunk. Two 16B loads:
//   elems 0..7  <- K = half*8 .. half*8+7
//   elems 8..15 <- K = 16+half*8 .. 16+half*8+7
static __device__ __forceinline__ v16bf load_tile(const bf16_t* base, int ld,
                                                  int row0, int k0, int lane) {
  int r   = row0 + (lane & 15);
  int sel = (lane >> 4) & 1;
  const bf16_t* p = base + (size_t)r * ld + k0 + sel * 8;
  v8bf lo = *(const v8bf*)(p);
  v8bf hi = *(const v8bf*)(p + 16);
  return __builtin_shufflevector(lo, hi, 0,1,2,3,4,5,6,7,8,9,10,11,12,13,14,15);
}

// CDNA5 async global->LDS copy, 16 bytes. Tracked by ASYNCcnt (not LOADcnt):
// no VGPR round-trip, drained with s_wait_asynccnt before the publishing
// barrier. lds_off is the LDS byte address (low 32 bits of generic address).
static __device__ __forceinline__ void async_ld16(unsigned lds_off,
                                                  const bf16_t* gptr) {
  asm volatile("global_load_async_to_lds_b128 %0, %1, off"
               :: "v"(lds_off), "v"(gptr) : "memory");
}
static __device__ __forceinline__ void wait_async_all() {
  asm volatile("s_wait_asynccnt 0x0" ::: "memory");
}
static __device__ __forceinline__ unsigned lds_addr(const void* p) {
  return (unsigned)(uintptr_t)p;   // addr[31:0] of generic LDS ptr = LDS offset
}

// ---------------------------------------------------------------------------
// Kernel 1: f32 -> bf16 convert (vector of 4 per thread)
// ---------------------------------------------------------------------------
struct alignas(8) bf16x4 { bf16_t v[4]; };

__global__ __launch_bounds__(256)
void k_cvt_bf16(const float4* __restrict__ x, bf16x4* __restrict__ o, int n4) {
  int i = blockIdx.x * blockDim.x + threadIdx.x;
  if (i < n4) {
    float4 f = x[i];
    bf16x4 r;
    r.v[0] = (bf16_t)f.x; r.v[1] = (bf16_t)f.y;
    r.v[2] = (bf16_t)f.z; r.v[3] = (bf16_t)f.w;
    o[i] = r;
  }
}

// ---------------------------------------------------------------------------
// Kernel 2: weight transpose + convert: wt[n*K+k] = (bf16) w[k*N+n]
// ---------------------------------------------------------------------------
__global__ __launch_bounds__(256)
void k_transpose_w(const float* __restrict__ w, bf16_t* __restrict__ wt,
                   int K, int N) {
  int i = blockIdx.x * blockDim.x + threadIdx.x;
  if (i < K * N) {
    int k = i / N, n = i % N;
    wt[(size_t)n * K + k] = (bf16_t)w[(size_t)k * N + n];
  }
}

// ---------------------------------------------------------------------------
// Kernel 3: fused QKV projection. 4 waves/WG; WG covers 128 rows x 64 cols.
// B-panel (64 N x 32 K) staged to LDS via async DMA, shared by all 4 waves;
// each wave owns a 32x64 output tile (8 WMMAs / K-step).
//   mode 0: Q -> [B][H][S][DK]   mode 1: K -> [B][H][S][DK]
//   mode 2: V -> transposed [B][H][DV][S]
// ---------------------------------------------------------------------------
__global__ __launch_bounds__(128)
void k_gemm_qkv(const bf16_t* __restrict__ xb,
                const bf16_t* __restrict__ wqt,
                const bf16_t* __restrict__ wkt,
                const bf16_t* __restrict__ wvt,
                bf16_t* __restrict__ qbuf,
                bf16_t* __restrict__ kbuf,
                bf16_t* __restrict__ vtb) {
  const int tid  = threadIdx.x;
  const int lane = tid & 31;
  const int wv   = tid >> 5;
  const int m0   = blockIdx.x * 128 + wv * 32;
  const int n0   = blockIdx.y * 64;
  const int mode = blockIdx.z;
  const bf16_t* wt = (mode == 0) ? wqt : (mode == 1) ? wkt : wvt;

  __shared__ __align__(16) bf16_t bpan[2][64 * 32];  // [n][k], ld=32

  // stage one 64x32 B panel (4KB): 256 x 16B chunks over 128 threads
  auto stage_b = [&](int buf, int k0) {
    unsigned base = lds_addr(&bpan[buf][0]);
    #pragma unroll
    for (int j = 0; j < 2; ++j) {
      int c = tid + j * 128;
      int r = c >> 2, q = c & 3;
      async_ld16(base + r * 64 + q * 16,
                 wt + (size_t)(n0 + r) * EMB + k0 + q * 8);
    }
  };

  v8f acc[2][4];
  #pragma unroll
  for (int mt = 0; mt < 2; ++mt)
    #pragma unroll
    for (int t = 0; t < 4; ++t) acc[mt][t] = (v8f)0.0f;

  stage_b(0, 0);
  wait_async_all();
  __syncthreads();

  for (int i = 0; i < EMB / 32; ++i) {
    const int k0 = i * 32, buf = i & 1;
    if (i + 1 < EMB / 32) stage_b(buf ^ 1, k0 + 32);

    v16bf a0 = load_tile(xb, EMB, m0,      k0, lane);
    v16bf a1 = load_tile(xb, EMB, m0 + 16, k0, lane);
    #pragma unroll
    for (int t = 0; t < 4; ++t) {
      v16bf bt = load_tile(&bpan[buf][0], 32, t * 16, 0, lane);
      acc[0][t] = wmma_bf16(a0, bt, acc[0][t]);
      acc[1][t] = wmma_bf16(a1, bt, acc[1][t]);
    }
    wait_async_all();
    __syncthreads();
  }

  const int half = lane >> 4;
  #pragma unroll
  for (int mt = 0; mt < 2; ++mt) {
    #pragma unroll
    for (int t = 0; t < 4; ++t) {
      #pragma unroll
      for (int v = 0; v < 8; ++v) {
        float val = acc[mt][t][v];
        int r   = m0 + mt * 16 + v + (half << 3);   // r = s*B + b
        int col = n0 + t * 16 + (lane & 15);        // col = h*64 + d
        int s = r >> 1, b = r & 1;
        int h = col >> 6, d = col & 63;
        size_t headbase = (size_t)(b * HEADS + h);
        if (mode == 0)
          qbuf[(headbase * S_LEN + s) * DKV + d] = (bf16_t)val;
        else if (mode == 1)
          kbuf[(headbase * S_LEN + s) * DKV + d] = (bf16_t)val;
        else
          vtb[(headbase * DKV + d) * S_LEN + s] = (bf16_t)val;
      }
    }
  }
}

// ---------------------------------------------------------------------------
// Kernel 4: flash attention. 4 waves/WG; WG = one (b,h) x 64 Q rows.
// K-panel (32x64) and V^T-panel (64x32) async-staged to LDS, double-buffered,
// shared by all 4 waves (4x L2 traffic reduction). Online softmax in C-layout
// registers; P relayout C->A via per-wave 1KB LDS tile.
// ---------------------------------------------------------------------------
__global__ __launch_bounds__(128)
void k_attention(const bf16_t* __restrict__ qbuf,
                 const bf16_t* __restrict__ kbuf,
                 const bf16_t* __restrict__ vtb,
                 bf16_t* __restrict__ ctx) {
  const int tid  = threadIdx.x;
  const int lane = tid & 31;
  const int wv   = tid >> 5;
  const int mgrp = blockIdx.x & (S_LEN / 64 - 1);   // 0..31
  const int bh   = blockIdx.x >> 5;                 // 0..31
  const int h    = bh & (HEADS - 1);
  const int b    = bh >> 4;
  const int s0   = mgrp * 64 + wv * 16;
  const int half = lane >> 4;

  const bf16_t* qh = qbuf + (size_t)bh * S_LEN * DKV;
  const bf16_t* kh = kbuf + (size_t)bh * S_LEN * DKV;
  const bf16_t* vh = vtb  + (size_t)bh * DKV * S_LEN;

  __shared__ __align__(16) bf16_t kpan[2][32 * 64];  // [key][dk],  ld=64
  __shared__ __align__(16) bf16_t vpan[2][64 * 32];  // [dv][key],  ld=32
  __shared__ __align__(16) bf16_t pbuf[4][16 * 32];  // per-wave P panel

  // stage K panel (4KB) + V^T panel (4KB): 2+2 chunks of 16B per thread
  auto stage_panel = [&](int buf, int key0) {
    unsigned kbase = lds_addr(&kpan[buf][0]);
    unsigned vbase = lds_addr(&vpan[buf][0]);
    #pragma unroll
    for (int j = 0; j < 2; ++j) {
      int c = tid + j * 128;
      int kr = c >> 3, kq = c & 7;
      async_ld16(kbase + kr * 128 + kq * 16,
                 kh + (size_t)(key0 + kr) * DKV + kq * 8);
      int vr = c >> 2, vq = c & 3;
      async_ld16(vbase + vr * 64 + vq * 16,
                 vh + (size_t)vr * S_LEN + key0 + vq * 8);
    }
  };

  // Q A-operand, K dim 64 = two 32-wide slabs (held for whole kernel)
  v16bf qa0 = load_tile(qh, DKV, s0, 0,  lane);
  v16bf qa1 = load_tile(qh, DKV, s0, 32, lane);

  v8f acc[4];
  #pragma unroll
  for (int t = 0; t < 4; ++t) acc[t] = (v8f)0.0f;
  float mrow[8], lrow[8];
  #pragma unroll
  for (int v = 0; v < 8; ++v) { mrow[v] = -3.0e38f; lrow[v] = 0.0f; }

  const float c = 0.125f * 1.44269504088896f;  // (1/sqrt(64)) * log2(e)
  bf16_t* pb = &pbuf[wv][0];

  stage_panel(0, 0);
  wait_async_all();
  __syncthreads();

  for (int kp = 0; kp < S_LEN / 32; ++kp) {
    const int buf = kp & 1;
    if (kp + 1 < S_LEN / 32) stage_panel(buf ^ 1, (kp + 1) * 32);
    const bf16_t* kl = &kpan[buf][0];
    const bf16_t* vl = &vpan[buf][0];

    // --- scores: two 16x16 tiles (local keys 0..15, 16..31) ---------------
    v8f z = (v8f)0.0f;
    v16bf kA0 = load_tile(kl, 64, 0,  0,  lane);
    v16bf kA1 = load_tile(kl, 64, 0,  32, lane);
    v8f sa = wmma_bf16(qa0, kA0, z);
    sa = wmma_bf16(qa1, kA1, sa);
    v16bf kB0 = load_tile(kl, 64, 16, 0,  lane);
    v16bf kB1 = load_tile(kl, 64, 16, 32, lane);
    v8f sb = wmma_bf16(qa0, kB0, z);
    sb = wmma_bf16(qa1, kB1, sb);

    // --- online softmax, per row (row = v + 8*half; spans 16 lanes) -------
    #pragma unroll
    for (int v = 0; v < 8; ++v) {
      float xa  = sa[v] * c;
      float xb2 = sb[v] * c;
      float t = fmaxf(xa, xb2);
      t = fmaxf(t, __shfl_xor(t, 1));
      t = fmaxf(t, __shfl_xor(t, 2));
      t = fmaxf(t, __shfl_xor(t, 4));
      t = fmaxf(t, __shfl_xor(t, 8));
      float mnew = fmaxf(mrow[v], t);
      float sc   = exp2f(mrow[v] - mnew);
      mrow[v] = mnew;
      float pa = exp2f(xa  - mnew);
      float pv = exp2f(xb2 - mnew);
      float rs = pa + pv;
      rs += __shfl_xor(rs, 1);
      rs += __shfl_xor(rs, 2);
      rs += __shfl_xor(rs, 4);
      rs += __shfl_xor(rs, 8);
      lrow[v] = lrow[v] * sc + rs;
      acc[0][v] *= sc; acc[1][v] *= sc; acc[2][v] *= sc; acc[3][v] *= sc;
      int m = v + (half << 3);
      int n = lane & 15;
      pb[m * 32 + n]      = (bf16_t)pa;
      pb[m * 32 + 16 + n] = (bf16_t)pv;
    }
    asm volatile("s_wait_dscnt 0x0" ::: "memory");

    // --- P(16x32) @ V^T tiles (N=DV, from LDS) ----------------------------
    v16bf pA = load_tile(pb, 32, 0, 0, lane);
    #pragma unroll
    for (int t = 0; t < 4; ++t) {
      v16bf vb = load_tile(vl, 32, t * 16, 0, lane);
      acc[t] = wmma_bf16(pA, vb, acc[t]);
    }

    wait_async_all();
    __syncthreads();
  }

  // --- epilogue: normalize and emit ctx as bf16 GEMM operand ---------------
  #pragma unroll
  for (int v = 0; v < 8; ++v) {
    float inv = 1.0f / lrow[v];
    int m = v + (half << 3);
    int srow = s0 + m;
    #pragma unroll
    for (int t = 0; t < 4; ++t) {
      int col = h * DKV + t * 16 + (lane & 15);
      ctx[(size_t)(srow * BATCH + b) * (HEADS * DKV) + col] =
          (bf16_t)(acc[t][v] * inv);
    }
  }
}

// ---------------------------------------------------------------------------
// Kernel 5: output projection ctx[4096,1024] @ Wo -> f32 out [S,B,E].
// Same 4-wave shared-B structure as k_gemm_qkv.
// ---------------------------------------------------------------------------
__global__ __launch_bounds__(128)
void k_gemm_out(const bf16_t* __restrict__ ctx,
                const bf16_t* __restrict__ wot,
                float* __restrict__ out) {
  const int tid  = threadIdx.x;
  const int lane = tid & 31;
  const int wv   = tid >> 5;
  const int m0   = blockIdx.x * 128 + wv * 32;
  const int n0   = blockIdx.y * 64;

  __shared__ __align__(16) bf16_t bpan[2][64 * 32];

  auto stage_b = [&](int buf, int k0) {
    unsigned base = lds_addr(&bpan[buf][0]);
    #pragma unroll
    for (int j = 0; j < 2; ++j) {
      int c = tid + j * 128;
      int r = c >> 2, q = c & 3;
      async_ld16(base + r * 64 + q * 16,
                 wot + (size_t)(n0 + r) * EMB + k0 + q * 8);
    }
  };

  v8f acc[2][4];
  #pragma unroll
  for (int mt = 0; mt < 2; ++mt)
    #pragma unroll
    for (int t = 0; t < 4; ++t) acc[mt][t] = (v8f)0.0f;

  stage_b(0, 0);
  wait_async_all();
  __syncthreads();

  for (int i = 0; i < EMB / 32; ++i) {
    const int k0 = i * 32, buf = i & 1;
    if (i + 1 < EMB / 32) stage_b(buf ^ 1, k0 + 32);

    v16bf a0 = load_tile(ctx, EMB, m0,      k0, lane);
    v16bf a1 = load_tile(ctx, EMB, m0 + 16, k0, lane);
    #pragma unroll
    for (int t = 0; t < 4; ++t) {
      v16bf bt = load_tile(&bpan[buf][0], 32, t * 16, 0, lane);
      acc[0][t] = wmma_bf16(a0, bt, acc[0][t]);
      acc[1][t] = wmma_bf16(a1, bt, acc[1][t]);
    }
    wait_async_all();
    __syncthreads();
  }

  const int half = lane >> 4;
  #pragma unroll
  for (int mt = 0; mt < 2; ++mt) {
    #pragma unroll
    for (int t = 0; t < 4; ++t) {
      #pragma unroll
      for (int v = 0; v < 8; ++v) {
        int r   = m0 + mt * 16 + v + (half << 3);   // r = s*B + b -> [S,B,E]
        int col = n0 + t * 16 + (lane & 15);
        out[(size_t)r * EMB + col] = acc[mt][t][v];
      }
    }
  }
}

// ---------------------------------------------------------------------------
extern "C" void kernel_launch(void* const* d_in, const int* in_sizes, int n_in,
                              void* d_out, int out_size, void* d_ws, size_t ws_size,
                              hipStream_t stream) {
  const float* X  = (const float*)d_in[0];   // [S,B,E]
  const float* Wq = (const float*)d_in[1];   // [E, H*DK]
  const float* Wk = (const float*)d_in[2];
  const float* Wv = (const float*)d_in[3];
  const float* Wo = (const float*)d_in[4];   // [H*DV, E]
  float* out = (float*)d_out;

  char* ws = (char*)d_ws;
  const size_t MB = 1u << 20;
  bf16_t* xb  = (bf16_t*)(ws + 0);
  bf16_t* wqt = (bf16_t*)(ws + 8 * MB);
  bf16_t* wkt = (bf16_t*)(ws + 10 * MB);
  bf16_t* wvt = (bf16_t*)(ws + 12 * MB);
  bf16_t* wot = (bf16_t*)(ws + 14 * MB);
  bf16_t* qb  = (bf16_t*)(ws + 16 * MB);
  bf16_t* kb  = (bf16_t*)(ws + 24 * MB);
  bf16_t* vtb = (bf16_t*)(ws + 32 * MB);
  bf16_t* ctx = (bf16_t*)(ws + 40 * MB);

  // 1) X -> bf16
  {
    int n4 = ROWS * EMB / 4;
    k_cvt_bf16<<<dim3(n4 / 256), dim3(256), 0, stream>>>(
        (const float4*)X, (bf16x4*)xb, n4);
  }
  // 2) weight transposes (all 1024x1024)
  {
    int n = EMB * EMB;
    dim3 g(n / 256), blk(256);
    k_transpose_w<<<g, blk, 0, stream>>>(Wq, wqt, EMB, EMB);
    k_transpose_w<<<g, blk, 0, stream>>>(Wk, wkt, EMB, EMB);
    k_transpose_w<<<g, blk, 0, stream>>>(Wv, wvt, EMB, EMB);
    k_transpose_w<<<g, blk, 0, stream>>>(Wo, wot, EMB, EMB);
  }
  // 3) fused QKV projection: 32 x 16 x 3 WGs of 128 threads
  k_gemm_qkv<<<dim3(ROWS / 128, EMB / 64, 3), dim3(128), 0, stream>>>(
      xb, wqt, wkt, wvt, qb, kb, vtb);
  // 4) flash attention: B*H*(S/64) = 1024 WGs of 128 threads
  k_attention<<<dim3(BATCH * HEADS * (S_LEN / 64)), dim3(128), 0, stream>>>(
      qb, kb, vtb, ctx);
  // 5) output projection -> f32 [S,B,E]
  k_gemm_out<<<dim3(ROWS / 128, EMB / 64), dim3(128), 0, stream>>>(ctx, wot, out);
}
